// PointTrans_Layer_33182917329083
// MI455X (gfx1250) — matrix operand order repaired and checked by reading, not compiled, not run on hardware
//
#include <hip/hip_runtime.h>

typedef __attribute__((ext_vector_type(16))) _Float16 v16h;
typedef __attribute__((ext_vector_type(8)))  _Float16 v8h;
typedef __attribute__((ext_vector_type(8)))  float    v8f;

#define LDSTR 40   // halfs per staged row: 80B stride -> 16B aligned chunks, conflict-free across 16 rows

// ---- WMMA fragment loaders (16-bit, 16x16x32) ----
// A-matrix 16x32 (MxK): lanes 0-15 hold K=0..7 & 16..23, lanes 16-31 hold K=8..15 & 24..31 (ISA 7.12.2)
__device__ __forceinline__ v16h load_fragA(const _Float16* lds, int row_base, int lane) {
  int r = row_base + (lane & 15);
  int koff = (lane >> 4) << 3;                       // 0 or 8
  const _Float16* p = lds + r * LDSTR + koff;
  v8h lo = *(const v8h*)(p);                         // K = koff .. koff+7
  v8h hi = *(const v8h*)(p + 16);                    // K = koff+16 .. koff+23
  return __builtin_shufflevector(lo, hi, 0,1,2,3,4,5,6,7,8,9,10,11,12,13,14,15);
}
// B-matrix 32x16 (KxN), staged as rows of W (B[k][n] = W[n][k] for y = x @ W^T):
// lanes 0-15 hold K=0..7 & 16..23, lanes 16-31 hold K=8..15 & 24..31
__device__ __forceinline__ v16h load_fragB(const _Float16* lds, int col_base, int lane) {
  int ncol = col_base + (lane & 15);
  int koff = (lane >> 4) << 3;                       // 0 or 8
  const _Float16* p = lds + ncol * LDSTR + koff;
  v8h lo = *(const v8h*)(p);
  v8h hi = *(const v8h*)(p + 16);
  return __builtin_shufflevector(lo, hi, 0,1,2,3,4,5,6,7,8,9,10,11,12,13,14,15);
}

// order-preserving float<->uint encoding for atomic segment-max
__device__ __forceinline__ unsigned enc_f32(float f) {
  unsigned b = __float_as_uint(f);
  return (b & 0x80000000u) ? ~b : (b | 0x80000000u);
}
__device__ __forceinline__ float dec_f32(unsigned u) {
  unsigned b = (u & 0x80000000u) ? (u & 0x7fffffffu) : ~u;
  return __uint_as_float(b);
}

// ---------------- K_init: zero L2-resident accumulators ----------------
__global__ void pt_init_kernel(unsigned* __restrict__ mmax, float* __restrict__ den,
                               float* __restrict__ acc, float* __restrict__ bnbuf,
                               long long n32) {
  long long i = (long long)blockIdx.x * blockDim.x + threadIdx.x;
  if (i < n32) { mmax[i] = 0u; den[i] = 0.f; acc[i] = 0.f; }
  if (i < 128) bnbuf[i] = 0.f;
}

// ---------------- K0: node projections a_src/a_dst/v via WMMA ----------------
__global__ void pt_node_proj_kernel(const float* __restrict__ x,
                                    const float* __restrict__ Wsrc,
                                    const float* __restrict__ Wdst,
                                    const float* __restrict__ Wval,
                                    float* __restrict__ a_src,
                                    float* __restrict__ a_dst,
                                    float* __restrict__ vproj,
                                    int n) {
  __shared__ _Float16 xt[64 * LDSTR];
  __shared__ _Float16 wt[3][32 * LDSTR];
  int tid = threadIdx.x;
  const float* Ws[3] = {Wsrc, Wdst, Wval};
  for (int i = tid; i < 3 * 1024; i += 256) {
    int m = i >> 10, rem = i & 1023;
    wt[m][(rem >> 5) * LDSTR + (rem & 31)] = (_Float16)Ws[m][rem];
  }
  int row0 = blockIdx.x * 64;
  const bool full = (row0 + 64 <= n);
  #pragma unroll
  for (int i = 0; i < 8; ++i) {
    int entry = tid + i * 256;
    int rl = entry >> 5, cc = entry & 31;
    int gr = row0 + rl;
    float val = (full || gr < n) ? x[(long long)gr * 32 + cc] : 0.f;
    xt[rl * LDSTR + cc] = (_Float16)val;
  }
  __syncthreads();
  int wave = tid >> 5, lane = tid & 31;
  int rt = wave & 3, ct = wave >> 2;
  v16h a = load_fragA(xt, rt * 16, lane);
  float* outs[3] = {a_src, a_dst, vproj};
  int col = ct * 16 + (lane & 15);
  int rbase = row0 + rt * 16 + ((lane >> 4) << 3);
  for (int m = 0; m < 3; ++m) {
    v16h b = load_fragB(wt[m], ct * 16, lane);
    v8f c = {};
    c = __builtin_amdgcn_wmma_f32_16x16x32_f16(false, a, false, b, (short)0, c, false, false);
    float* o = outs[m] + (long long)rbase * 32 + col;
    if (full) {
      #pragma unroll
      for (int g = 0; g < 8; ++g) o[g * 32] = c[g];
    } else {
      #pragma unroll
      for (int g = 0; g < 8; ++g)
        if (rbase + g < n) o[g * 32] = c[g];
    }
  }
}

// ---------------- K1: edge h = (a_dst[d]-a_src[s]+delta) @ Wa1^T + ba1; BN stats ----------------
__global__ void pt_edge_h_kernel(const long long* __restrict__ ei,
                                 const float* __restrict__ pos,
                                 const float* __restrict__ a_src,
                                 const float* __restrict__ a_dst,
                                 const float* __restrict__ Wa1,
                                 const float* __restrict__ ba1,
                                 const float* __restrict__ Wp,
                                 const float* __restrict__ bp,
                                 float* __restrict__ hbuf,
                                 float* __restrict__ bn_sum,
                                 float* __restrict__ bn_sumsq,
                                 long long ecount) {
  __shared__ _Float16 tt[64 * LDSTR];
  __shared__ _Float16 w1h[32 * LDSTR];
  __shared__ float wp_l[96], bp_l[32], bns[32], bns2[32];
  __shared__ int ssl[64], ddl[64];
  __shared__ float p0l[64], p1l[64], p2l[64];
  int tid = threadIdx.x;
  long long e0 = (long long)blockIdx.x * 64;
  __builtin_prefetch(ei + e0 + 64, 0, 0);            // global_prefetch_b8: next edge tile
  for (int i = tid; i < 1024; i += 256)
    w1h[(i >> 5) * LDSTR + (i & 31)] = (_Float16)Wa1[i];
  if (tid < 96) wp_l[tid] = Wp[tid];
  if (tid < 32) { bp_l[tid] = bp[tid]; bns[tid] = 0.f; bns2[tid] = 0.f; }
  if (tid < 64) {                                     // stage per-edge indices + pos delta once
    long long e = e0 + tid;
    long long s = 0, d = 0;
    if (e < ecount) { s = ei[e]; d = ei[ecount + e]; }
    ssl[tid] = (int)s; ddl[tid] = (int)d;
    p0l[tid] = pos[d * 3 + 0] - pos[s * 3 + 0];
    p1l[tid] = pos[d * 3 + 1] - pos[s * 3 + 1];
    p2l[tid] = pos[d * 3 + 2] - pos[s * 3 + 2];
  }
  __syncthreads();
  #pragma unroll
  for (int i = 0; i < 8; ++i) {
    int entry = tid + i * 256;
    int el = entry >> 5, cc = entry & 31;
    long long s = ssl[el], d = ddl[el];
    float del = wp_l[cc * 3] * p0l[el] + wp_l[cc * 3 + 1] * p1l[el] +
                wp_l[cc * 3 + 2] * p2l[el] + bp_l[cc];
    float tv = a_dst[d * 32 + cc] - a_src[s * 32 + cc] + del;
    tt[el * LDSTR + cc] = (_Float16)tv;
  }
  __syncthreads();
  int wave = tid >> 5, lane = tid & 31;
  int rt = wave & 3, ct = wave >> 2;
  v16h a = load_fragA(tt, rt * 16, lane);
  v16h b = load_fragB(w1h, ct * 16, lane);
  int col = ct * 16 + (lane & 15);
  float bias = ba1[col];
  v8f c;
  #pragma unroll
  for (int g = 0; g < 8; ++g) c[g] = bias;
  c = __builtin_amdgcn_wmma_f32_16x16x32_f16(false, a, false, b, (short)0, c, false, false);
  long long rbase = e0 + rt * 16 + ((lane >> 4) << 3);
  float* o = hbuf + rbase * 32 + col;
  float s1 = 0.f, s2 = 0.f;
  const bool full = (e0 + 64 <= ecount);
  if (full) {
    #pragma unroll
    for (int g = 0; g < 8; ++g) {
      o[g * 32] = c[g];
      s1 += c[g]; s2 += c[g] * c[g];
    }
  } else {
    #pragma unroll
    for (int g = 0; g < 8; ++g) {
      if (rbase + g < ecount) { o[g * 32] = c[g]; s1 += c[g]; s2 += c[g] * c[g]; }
    }
  }
  s1 += __shfl_down(s1, 16);                          // combine half-waves (same channel)
  s2 += __shfl_down(s2, 16);
  if (lane < 16) { atomicAdd(&bns[col], s1); atomicAdd(&bns2[col], s2); }
  __syncthreads();
  if (tid < 32) { atomicAdd(&bn_sum[tid], bns[tid]); atomicAdd(&bn_sumsq[tid], bns2[tid]); }
}

// ---------------- K2: fold BN into per-channel affine ----------------
__global__ void pt_bn_finalize_kernel(const float* __restrict__ bn_sum,
                                      const float* __restrict__ bn_sumsq,
                                      const float* __restrict__ gamma,
                                      const float* __restrict__ beta,
                                      float* __restrict__ scale,
                                      float* __restrict__ shift, float inv_e) {
  int c = threadIdx.x;
  if (c < 32) {
    float mu = bn_sum[c] * inv_e;
    float var = bn_sumsq[c] * inv_e - mu * mu;
    float rs = rsqrtf(var + 1e-5f);
    float sc = gamma[c] * rs;
    scale[c] = sc;
    shift[c] = beta[c] - mu * sc;
  }
}

// ---------------- K3: logits = relu(BN(h)) @ Wa2^T + ba2 (in place); segment max ----------------
__global__ void pt_edge_logits_kernel(const long long* __restrict__ ei,
                                      const float* __restrict__ Wa2,
                                      const float* __restrict__ ba2,
                                      const float* __restrict__ scale,
                                      const float* __restrict__ shift,
                                      float* __restrict__ hbuf,
                                      unsigned* __restrict__ mmax,
                                      long long ecount) {
  __shared__ _Float16 tt[64 * LDSTR];
  __shared__ _Float16 w2h[32 * LDSTR];
  __shared__ float sc_l[32], sh_l[32];
  __shared__ int ddl[64];
  int tid = threadIdx.x;
  long long e0 = (long long)blockIdx.x * 64;
  for (int i = tid; i < 1024; i += 256)
    w2h[(i >> 5) * LDSTR + (i & 31)] = (_Float16)Wa2[i];
  if (tid < 32) { sc_l[tid] = scale[tid]; sh_l[tid] = shift[tid]; }
  if (tid < 64) {
    long long e = e0 + tid;
    ddl[tid] = (e < ecount) ? (int)ei[ecount + e] : 0;
  }
  __syncthreads();
  const bool full = (e0 + 64 <= ecount);
  #pragma unroll
  for (int i = 0; i < 8; ++i) {
    int entry = tid + i * 256;
    int el = entry >> 5, cc = entry & 31;
    long long idx = e0 * 32 + entry;                  // fully coalesced streaming read
    float hv = (full || idx < ecount * 32) ? hbuf[idx] : 0.f;
    hv = fmaxf(hv * sc_l[cc] + sh_l[cc], 0.f);
    tt[el * LDSTR + cc] = (_Float16)hv;
  }
  __syncthreads();                                    // all reads done before in-place writes
  int wave = tid >> 5, lane = tid & 31;
  int rt = wave & 3, ct = wave >> 2;
  v16h a = load_fragA(tt, rt * 16, lane);
  v16h b = load_fragB(w2h, ct * 16, lane);
  int col = ct * 16 + (lane & 15);
  float bias = ba2[col];
  v8f c;
  #pragma unroll
  for (int g = 0; g < 8; ++g) c[g] = bias;
  c = __builtin_amdgcn_wmma_f32_16x16x32_f16(false, a, false, b, (short)0, c, false, false);
  int lrow = rt * 16 + ((lane >> 4) << 3);            // local row in 64-edge tile
  float* o = hbuf + (e0 + lrow) * 32 + col;
  if (full) {
    #pragma unroll
    for (int g = 0; g < 8; ++g) {
      float L = c[g];
      o[g * 32] = L;
      long long d = (long long)ddl[lrow + g];
      atomicMax(&mmax[d * 32 + col], enc_f32(L));     // L2-resident segment max
    }
  } else {
    #pragma unroll
    for (int g = 0; g < 8; ++g) {
      if (e0 + lrow + g < ecount) {
        float L = c[g];
        o[g * 32] = L;
        long long d = (long long)ddl[lrow + g];
        atomicMax(&mmax[d * 32 + col], enc_f32(L));
      }
    }
  }
}

// ---------------- K4: ex = exp(logit - m[dst]) in place; den[dst] += ex ----------------
__global__ void pt_softmax_den_kernel(const long long* __restrict__ ei,
                                      float* __restrict__ hbuf,
                                      const unsigned* __restrict__ mmax,
                                      float* __restrict__ den,
                                      long long ecount) {
  __shared__ int ddl[32];
  int tid = threadIdx.x;
  long long e0 = (long long)blockIdx.x * 32;          // 32 edges x 32 ch = 1024 entries
  if (tid < 32) {
    long long e = e0 + tid;
    ddl[tid] = (e < ecount) ? (int)ei[ecount + e] : 0;
  }
  __syncthreads();
  const bool full = (e0 + 32 <= ecount);
  __builtin_prefetch(hbuf + (e0 + 32) * 32, 0, 0);
  #pragma unroll
  for (int i = 0; i < 4; ++i) {
    int entry = tid + i * 256;
    int el = entry >> 5, cc = entry & 31;
    long long idx = e0 * 32 + entry;
    if (full || idx < ecount * 32) {
      long long d = (long long)ddl[el];
      float L = hbuf[idx];
      float m = dec_f32(mmax[d * 32 + cc]);
      float ex = __expf(L - m);
      hbuf[idx] = ex;
      atomicAdd(&den[d * 32 + cc], ex);
    }
  }
}

// ---------------- K5: acc[dst] += ex/den * (v[src] + delta) ----------------
__global__ void pt_scatter_kernel(const long long* __restrict__ ei,
                                  const float* __restrict__ pos,
                                  const float* __restrict__ vproj,
                                  const float* __restrict__ Wp,
                                  const float* __restrict__ bp,
                                  const float* __restrict__ hbuf,
                                  const float* __restrict__ den,
                                  float* __restrict__ acc,
                                  long long ecount) {
  __shared__ float wp_l[96], bp_l[32];
  __shared__ int ssl[64], ddl[64];
  __shared__ float p0l[64], p1l[64], p2l[64];
  int tid = threadIdx.x;
  long long e0 = (long long)blockIdx.x * 64;
  __builtin_prefetch(hbuf + (e0 + 64) * 32, 0, 0);
  if (tid < 96) wp_l[tid] = Wp[tid];
  if (tid < 32) bp_l[tid] = bp[tid];
  if (tid < 64) {
    long long e = e0 + tid;
    long long s = 0, d = 0;
    if (e < ecount) { s = ei[e]; d = ei[ecount + e]; }
    ssl[tid] = (int)s; ddl[tid] = (int)d;
    p0l[tid] = pos[d * 3 + 0] - pos[s * 3 + 0];
    p1l[tid] = pos[d * 3 + 1] - pos[s * 3 + 1];
    p2l[tid] = pos[d * 3 + 2] - pos[s * 3 + 2];
  }
  __syncthreads();
  const bool full = (e0 + 64 <= ecount);
  #pragma unroll
  for (int i = 0; i < 8; ++i) {
    int entry = tid + i * 256;
    int el = entry >> 5, cc = entry & 31;
    long long e = e0 + el;
    if (full || e < ecount) {
      long long s = ssl[el], d = ddl[el];
      float del = wp_l[cc * 3] * p0l[el] + wp_l[cc * 3 + 1] * p1l[el] +
                  wp_l[cc * 3 + 2] * p2l[el] + bp_l[cc];
      float ex = hbuf[e * 32 + cc];
      float alpha = ex / (den[d * 32 + cc] + 1e-16f);
      atomicAdd(&acc[d * 32 + cc], alpha * (vproj[s * 32 + cc] + del));
    }
  }
}

// ---------------- K6: out = acc @ Wu^T + bu + x ----------------
__global__ void pt_output_kernel(const float* __restrict__ acc,
                                 const float* __restrict__ x,
                                 const float* __restrict__ Wu,
                                 const float* __restrict__ bu,
                                 float* __restrict__ out, int n) {
  __shared__ _Float16 at[64 * LDSTR];
  __shared__ _Float16 wuh[32 * LDSTR];
  int tid = threadIdx.x;
  for (int i = tid; i < 1024; i += 256)
    wuh[(i >> 5) * LDSTR + (i & 31)] = (_Float16)Wu[i];
  int row0 = blockIdx.x * 64;
  const bool full = (row0 + 64 <= n);
  #pragma unroll
  for (int i = 0; i < 8; ++i) {
    int entry = tid + i * 256;
    int rl = entry >> 5, cc = entry & 31;
    int gr = row0 + rl;
    float val = (full || gr < n) ? acc[(long long)gr * 32 + cc] : 0.f;
    at[rl * LDSTR + cc] = (_Float16)val;
  }
  __syncthreads();
  int wave = tid >> 5, lane = tid & 31;
  int rt = wave & 3, ct = wave >> 2;
  v16h a = load_fragA(at, rt * 16, lane);
  v16h b = load_fragB(wuh, ct * 16, lane);
  int col = ct * 16 + (lane & 15);
  float bias = bu[col];
  v8f c;
  #pragma unroll
  for (int g = 0; g < 8; ++g) c[g] = bias;
  c = __builtin_amdgcn_wmma_f32_16x16x32_f16(false, a, false, b, (short)0, c, false, false);
  int rbase = row0 + rt * 16 + ((lane >> 4) << 3);
  const float* xi = x + (long long)rbase * 32 + col;
  float* o = out + (long long)rbase * 32 + col;
  if (full) {
    #pragma unroll
    for (int g = 0; g < 8; ++g) o[g * 32] = c[g] + xi[g * 32];
  } else {
    #pragma unroll
    for (int g = 0; g < 8; ++g)
      if (rbase + g < n) o[g * 32] = c[g] + xi[g * 32];
  }
}

extern "C" void kernel_launch(void* const* d_in, const int* in_sizes, int n_in,
                              void* d_out, int out_size, void* d_ws, size_t ws_size,
                              hipStream_t stream) {
  const float*     x   = (const float*)d_in[0];
  const float*     pos = (const float*)d_in[1];
  const long long* ei  = (const long long*)d_in[2];   // int64 [2,E]
  const float* W_src = (const float*)d_in[3];
  const float* W_dst = (const float*)d_in[4];
  const float* W_val = (const float*)d_in[5];
  const float* Wp    = (const float*)d_in[6];
  const float* bp    = (const float*)d_in[7];
  const float* Wa1   = (const float*)d_in[8];
  const float* ba1   = (const float*)d_in[9];
  const float* gamma = (const float*)d_in[10];
  const float* beta  = (const float*)d_in[11];
  const float* Wa2   = (const float*)d_in[12];
  const float* ba2   = (const float*)d_in[13];
  const float* Wu    = (const float*)d_in[14];
  const float* bu    = (const float*)d_in[15];
  float* out = (float*)d_out;

  const int n = in_sizes[0] / 32;
  const long long e = (long long)(in_sizes[2] / 2);
  const long long n32 = (long long)n * 32;

  // workspace layout (f32 units): a_src | a_dst | v | mmax(u32) | den | acc | bn(128) | hbuf
  float* ws     = (float*)d_ws;
  float* a_src  = ws;
  float* a_dst  = a_src + n32;
  float* vproj  = a_dst + n32;
  unsigned* mmax = (unsigned*)(vproj + n32);
  float* den    = (float*)(mmax + n32);
  float* acc    = den + n32;
  float* bnbuf  = acc + n32;          // [0..31] sum, [32..63] sumsq, [64..95] scale, [96..127] shift
  float* bn_sum = bnbuf, *bn_sumsq = bnbuf + 32, *scale = bnbuf + 64, *shift = bnbuf + 96;
  float* hbuf   = bnbuf + 128;        // [E,32] h -> logits -> ex (in place)

  const int nblk   = (n + 63) / 64;
  const int eblk   = (int)((e + 63) / 64);
  const int eblk32 = (int)((e + 31) / 32);

  pt_init_kernel<<<(int)((n32 + 255) / 256), 256, 0, stream>>>(mmax, den, acc, bnbuf, n32);
  pt_node_proj_kernel<<<nblk, 256, 0, stream>>>(x, W_src, W_dst, W_val, a_src, a_dst, vproj, n);
  pt_edge_h_kernel<<<eblk, 256, 0, stream>>>(ei, pos, a_src, a_dst, Wa1, ba1, Wp, bp,
                                             hbuf, bn_sum, bn_sumsq, e);
  pt_bn_finalize_kernel<<<1, 32, 0, stream>>>(bn_sum, bn_sumsq, gamma, beta, scale, shift,
                                              1.0f / (float)e);
  pt_edge_logits_kernel<<<eblk, 256, 0, stream>>>(ei, Wa2, ba2, scale, shift, hbuf, mmax, e);
  pt_softmax_den_kernel<<<eblk32, 256, 0, stream>>>(ei, hbuf, mmax, den, e);
  pt_scatter_kernel<<<eblk, 256, 0, stream>>>(ei, pos, vproj, Wp, bp, hbuf, den, acc, e);
  pt_output_kernel<<<nblk, 256, 0, stream>>>(acc, x, Wu, bu, out, n);
}
